// LSDABlock_45810121179567
// MI455X (gfx1250) — matrix operand order repaired and checked by reading, not compiled, not run on hardware
//
#include <hip/hip_runtime.h>

#define DEVI __device__ __forceinline__

typedef __attribute__((ext_vector_type(16))) _Float16 v16h;
typedef __attribute__((ext_vector_type(8)))  _Float16 v8h;
typedef __attribute__((ext_vector_type(8)))  float    v8f;
typedef __attribute__((ext_vector_type(4)))  unsigned int v4u;
typedef __attribute__((ext_vector_type(8)))  int v8i;
typedef __attribute__((ext_vector_type(4)))  int v4i;

union V16U { v16h v; v8h h[2]; };

constexpr int Bn = 32, R = 64, L = R * R, C = 96, NH = 3, HD = 32, G = 8;
constexpr int WIN = G * G;            // 64 tokens per window
constexpr int HID = 4 * C;            // 384
constexpr int NWIN = Bn * (R / G) * (R / G);   // 2048 windows
constexpr int MROWS = Bn * L;                  // 131072 tokens
constexpr float SCALE = 0.17677669529663687f;  // HD^-0.5

// ---- workspace layout (bytes), 256B aligned chunks ----
constexpr size_t ALGN(size_t x) { return (x + 255) & ~size_t(255); }
constexpr size_t OFF_WQKVT = 0;                                          // [288][96] f16 (B^T)
constexpr size_t OFF_WPROJT = ALGN(OFF_WQKVT + 288ull * 96 * 2);         // [96][96] f16
constexpr size_t OFF_WFC1T  = ALGN(OFF_WPROJT + 96ull * 96 * 2);         // [384][96] f16
constexpr size_t OFF_WFC2T  = ALGN(OFF_WFC1T + 384ull * 96 * 2);         // [96][384] f16
constexpr size_t OFF_HWIN   = ALGN(OFF_WFC2T + 96ull * 384 * 2);         // [MROWS][96] f16 windowed LN1
constexpr size_t OFF_Q      = ALGN(OFF_HWIN + (size_t)MROWS * C * 2);    // [NWIN][NH][64][32] f16 (pre-scaled)
constexpr size_t OFF_K      = ALGN(OFF_Q + (size_t)NWIN * NH * WIN * HD * 2);
constexpr size_t OFF_VT     = ALGN(OFF_K + (size_t)NWIN * NH * WIN * HD * 2); // [NWIN][NH][32][64] f16
constexpr size_t OFF_AO     = ALGN(OFF_VT + (size_t)NWIN * NH * HD * WIN * 2); // [MROWS][96] f16 windowed attn-out
constexpr size_t OFF_X1     = ALGN(OFF_AO + (size_t)MROWS * C * 2);      // [MROWS][96] f32 token-major
constexpr size_t OFF_H2     = ALGN(OFF_X1 + (size_t)MROWS * C * 4);      // [MROWS][96] f16 LN2 out

// ---- WMMA helper (CDNA5 v_wmma_f32_16x16x32_f16, wave32) ----
DEVI v8f wmma_f16(v16h a, v16h b, v8f c) {
  return __builtin_amdgcn_wmma_f32_16x16x32_f16(false, a, false, b, (short)0, c, false, false);
}

// A tile 16x32 from row-major f16 matrix; p16 points at (row0, k0); ld in halves.
DEVI v16h load_a(const _Float16* p16, int ld, int lane) {
  int hf = lane >> 4, r = lane & 15;
  const _Float16* p = p16 + (size_t)r * ld + hf * 8;
  V16U u;
  u.h[0] = *(const v8h*)(p);
  u.h[1] = *(const v8h*)(p + 16);
  return u.v;
}
// B tile 32x16 from B-transposed row-major (ldT halves); pT points at (n0, k0) of B^T.
DEVI v16h load_b(const _Float16* pT, int ldT, int lane) {
  int hf = lane >> 4, n = lane & 15;
  const _Float16* p = pT + (size_t)n * ldT + hf * 8;
  V16U u;
  u.h[0] = *(const v8h*)(p);
  u.h[1] = *(const v8h*)(p + 16);
  return u.v;
}

// ---- Tensor Data Mover: DMA a 2-D f16 panel (rows x rowElems, packed) global->LDS ----
// D# per CDNA5 ISA ch.8: group0 = {flags/count, lds_addr, global_addr, type=2},
// group1 = {data_size=2B, tensor_dim0/1, tile_dim0/1, dim0 stride}; groups 2/3 unused (2-D).
// This toolchain's builtin takes 6 args (g0 v4u, g1 v8i, g2 v4i, g3 v4i, extra v8i, cpol).
DEVI void tdm_load_panel_f16(void* ldsDst, const void* gsrc, unsigned rowElems, unsigned rows) {
  unsigned lds_off = (unsigned)(uintptr_t)ldsDst;               // flat low 32 bits == LDS byte address
  unsigned long long ga = (unsigned long long)(uintptr_t)gsrc;
  v4u g0 = {};
  g0[0] = 1u;                                                   // count=1, user descriptor
  g0[1] = lds_off;                                              // lds_addr
  g0[2] = (unsigned)(ga & 0xffffffffu);                         // global_addr[31:0]
  g0[3] = (unsigned)((ga >> 32) & 0x01ffffffu) | (2u << 30);    // global_addr[56:32] | type=2
  v8i g1 = {};
  g1[0] = (int)(1u << 16);                                      // data_size=1 -> 2 bytes
  g1[1] = (int)((rowElems & 0xffffu) << 16);                    // tensor_dim0[15:0]
  g1[2] = (int)(((rowElems >> 16) & 0xffffu) | ((rows & 0xffffu) << 16));  // td0 hi | td1 lo
  g1[3] = (int)(((rows >> 16) & 0xffffu) | ((rowElems & 0xffffu) << 16));  // td1 hi | tile_dim0
  g1[4] = (int)(rows & 0xffffu);                                // tile_dim1 (tile_dim2=0)
  g1[5] = (int)rowElems;                                        // tensor_dim0_stride[31:0]
  v4i z4 = {};
  v8i z8 = {};
  __builtin_amdgcn_tensor_load_to_lds(g0, g1, z4, z4, z8, 0);
}

// ---- kernel 0: convert + transpose all weights to f16 B^T form ----
__global__ void k_cvtw(const float* __restrict__ wqkv, const float* __restrict__ wproj,
                       const float* __restrict__ wfc1, const float* __restrict__ wfc2,
                       char* __restrict__ ws) {
  _Float16* wqkvT = (_Float16*)(ws + OFF_WQKVT);
  _Float16* wprojT = (_Float16*)(ws + OFF_WPROJT);
  _Float16* wfc1T = (_Float16*)(ws + OFF_WFC1T);
  _Float16* wfc2T = (_Float16*)(ws + OFF_WFC2T);
  int i = blockIdx.x * blockDim.x + threadIdx.x;
  if (i < 288 * 96) { int n = i / 96, k = i % 96; wqkvT[n * 96 + k] = (_Float16)wqkv[k * 288 + n]; return; }
  int j = i - 288 * 96;
  if (j < 96 * 96) { int n = j / 96, k = j % 96; wprojT[n * 96 + k] = (_Float16)wproj[k * 96 + n]; return; }
  int p = j - 96 * 96;
  if (p < 384 * 96) { int n = p / 96, k = p % 96; wfc1T[n * 96 + k] = (_Float16)wfc1[k * 384 + n]; return; }
  int q = p - 384 * 96;
  if (q < 96 * 384) { int n = q / 384, k = q % 384; wfc2T[n * 384 + k] = (_Float16)wfc2[k * 96 + n]; }
}

// ---- kernel 1: LN1 + window partition -> hwin f16 (wave per token) ----
__global__ void k_ln1(const float* __restrict__ x, const float* __restrict__ g,
                      const float* __restrict__ bt, char* __restrict__ ws) {
  _Float16* hwin = (_Float16*)(ws + OFF_HWIN);
  int wave = (blockIdx.x * blockDim.x + threadIdx.x) >> 5;
  int lane = threadIdx.x & 31;
  if (wave >= MROWS) return;
  const float* xr = x + (size_t)wave * C;
  float v0 = xr[lane], v1 = xr[lane + 32], v2 = xr[lane + 64];
  float s = v0 + v1 + v2;
#pragma unroll
  for (int m = 16; m >= 1; m >>= 1) s += __shfl_xor(s, m, 32);
  float mean = s * (1.0f / C);
  float d0 = v0 - mean, d1 = v1 - mean, d2 = v2 - mean;
  float q = d0 * d0 + d1 * d1 + d2 * d2;
#pragma unroll
  for (int m = 16; m >= 1; m >>= 1) q += __shfl_xor(q, m, 32);
  float rstd = rsqrtf(q * (1.0f / C) + 1e-5f);
  int b = wave >> 12, l = wave & 4095;
  int yy = l >> 6, xx = l & 63;
  int win = (b << 6) + ((yy >> 3) << 3) + (xx >> 3);
  int t = ((yy & 7) << 3) + (xx & 7);
  _Float16* hr = hwin + ((size_t)win * WIN + t) * C;
  hr[lane]      = (_Float16)(d0 * rstd * g[lane]      + bt[lane]);
  hr[lane + 32] = (_Float16)(d1 * rstd * g[lane + 32] + bt[lane + 32]);
  hr[lane + 64] = (_Float16)(d2 * rstd * g[lane + 64] + bt[lane + 64]);
}

// ---- kernel 2: QKV GEMM (M=131072, K=96, N=288); weights TDM-staged in LDS ----
__global__ void __launch_bounds__(256) k_qkv(const float* __restrict__ bqkv,
                                             char* __restrict__ ws) {
  extern __shared__ __align__(16) char smem[];
  _Float16* wLds = (_Float16*)smem;                 // [288][96] f16
  if (threadIdx.x < 32) {                           // one TDM issue per block (wave 0)
    tdm_load_panel_f16(smem, ws + OFF_WQKVT, 96, 288);
    __builtin_amdgcn_s_wait_tensorcnt(0);
  }
  __syncthreads();
  const _Float16* hwin = (const _Float16*)(ws + OFF_HWIN);
  _Float16* qb = (_Float16*)(ws + OFF_Q);
  _Float16* kb = (_Float16*)(ws + OFF_K);
  _Float16* vT = (_Float16*)(ws + OFF_VT);
  int wave = (blockIdx.x * blockDim.x + threadIdx.x) >> 5;
  int lane = threadIdx.x & 31;
  int row0 = wave * 16;
  int hf = lane >> 4, col = lane & 15;
  v16h a0 = load_a(hwin + (size_t)row0 * C + 0, C, lane);
  v16h a1 = load_a(hwin + (size_t)row0 * C + 32, C, lane);
  v16h a2 = load_a(hwin + (size_t)row0 * C + 64, C, lane);
  for (int nt = 0; nt < 18; ++nt) {
    int n0 = nt * 16;
    v16h b0 = load_b(wLds + (size_t)n0 * C + 0, C, lane);
    v16h b1 = load_b(wLds + (size_t)n0 * C + 32, C, lane);
    v16h b2 = load_b(wLds + (size_t)n0 * C + 64, C, lane);
    v8f acc = {};
    acc = wmma_f16(a0, b0, acc);
    acc = wmma_f16(a1, b1, acc);
    acc = wmma_f16(a2, b2, acc);
    int j = n0 + col;
    float bias = bqkv[j];
    int sect = j / C, jc = j % C;
    int head = jc >> 5, hd = jc & 31;
#pragma unroll
    for (int r = 0; r < 8; ++r) {
      int row = row0 + r + hf * 8;
      int win = row >> 6, t = row & 63;
      float val = acc[r] + bias;
      size_t hb = (size_t)win * NH + head;
      if (sect == 0)      qb[(hb * WIN + t) * HD + hd] = (_Float16)(val * SCALE);
      else if (sect == 1) kb[(hb * WIN + t) * HD + hd] = (_Float16)val;
      else                vT[(hb * HD + hd) * WIN + t] = (_Float16)val;
    }
  }
}

// ---- kernel 3: attention per (window, head): 4 waves, each 16 rows ----
__global__ void __launch_bounds__(128) k_attn(const float* __restrict__ wle,
                                              const float* __restrict__ ble,
                                              float* __restrict__ outAttn,
                                              char* __restrict__ ws) {
  __shared__ __align__(16) _Float16 pbuf[4][16 * 64];
  const _Float16* qb = (const _Float16*)(ws + OFF_Q);
  const _Float16* kb = (const _Float16*)(ws + OFF_K);
  const _Float16* vT = (const _Float16*)(ws + OFF_VT);
  _Float16* ao = (_Float16*)(ws + OFF_AO);
  int win = blockIdx.x / NH, head = blockIdx.x % NH;
  int w = threadIdx.x >> 5, lane = threadIdx.x & 31;
  int hf = lane >> 4, col = lane & 15;
  int r0 = w * 16;
  size_t qkbase = ((size_t)win * NH + head) * WIN * HD;
  v16h a = load_a(qb + qkbase + (size_t)r0 * HD, HD, lane);
  v8f s[4];
#pragma unroll
  for (int nt = 0; nt < 4; ++nt) {
    v8f z = {};
    s[nt] = wmma_f16(a, load_b(kb + qkbase + (size_t)nt * 16 * HD, HD, lane), z);
  }
  // softmax: row m = r + 8*(lane>=16); row spans 16 lanes of its half x 4 tiles
  float mx[8], sm[8];
#pragma unroll
  for (int r = 0; r < 8; ++r) {
    float m = fmaxf(fmaxf(s[0][r], s[1][r]), fmaxf(s[2][r], s[3][r]));
#pragma unroll
    for (int k = 8; k >= 1; k >>= 1) m = fmaxf(m, __shfl_xor(m, k, 32));
    mx[r] = m;
  }
#pragma unroll
  for (int nt = 0; nt < 4; ++nt)
#pragma unroll
    for (int r = 0; r < 8; ++r) s[nt][r] = __expf(s[nt][r] - mx[r]);
#pragma unroll
  for (int r = 0; r < 8; ++r) {
    float t = s[0][r] + s[1][r] + s[2][r] + s[3][r];
#pragma unroll
    for (int k = 8; k >= 1; k >>= 1) t += __shfl_xor(t, k, 32);
    sm[r] = 1.0f / t;
  }
  size_t abase = ((size_t)win * NH + head) * WIN * WIN;
#pragma unroll
  for (int nt = 0; nt < 4; ++nt) {
#pragma unroll
    for (int r = 0; r < 8; ++r) {
      float p = s[nt][r] * sm[r];
      int m = r + hf * 8;
      outAttn[abase + (size_t)(r0 + m) * WIN + nt * 16 + col] = p;
      pbuf[w][m * 64 + nt * 16 + col] = (_Float16)p;
    }
  }
  __syncthreads();
  v16h pa0 = load_a(&pbuf[w][0], 64, lane);
  v16h pa1 = load_a(&pbuf[w][32], 64, lane);
  size_t vbase = ((size_t)win * NH + head) * HD * WIN;
  v8f o[2];
#pragma unroll
  for (int nt = 0; nt < 2; ++nt) {
    v8f z = {};
    z = wmma_f16(pa0, load_b(vT + vbase + (size_t)nt * 16 * WIN + 0, WIN, lane), z);
    o[nt] = wmma_f16(pa1, load_b(vT + vbase + (size_t)nt * 16 * WIN + 32, WIN, lane), z);
  }
  // LePE depthwise 3x3 on vT image (channel-major [hd][8x8]) + store windowed attn-out
#pragma unroll
  for (int nt = 0; nt < 2; ++nt) {
    int hd = nt * 16 + col;
    int c = head * HD + hd;
    float wk[9];
#pragma unroll
    for (int i = 0; i < 9; ++i) wk[i] = wle[c * 9 + i];
    float bl = ble[c];
    const _Float16* vch = vT + vbase + (size_t)hd * WIN;
#pragma unroll
    for (int r = 0; r < 8; ++r) {
      int tok = r0 + r + hf * 8;
      int y = tok >> 3, xq = tok & 7;
      float acc = bl;
#pragma unroll
      for (int dy = -1; dy <= 1; ++dy) {
        int yy = y + dy;
        if ((unsigned)yy >= 8u) continue;
#pragma unroll
        for (int dx = -1; dx <= 1; ++dx) {
          int xx = xq + dx;
          if ((unsigned)xx >= 8u) continue;
          acc += wk[(dy + 1) * 3 + dx + 1] * (float)vch[yy * 8 + xx];
        }
      }
      ao[((size_t)win * WIN + tok) * C + c] = (_Float16)(o[nt][r] + acc);
    }
  }
}

// ---- kernel 4: proj GEMM + window reverse + shortcut -> x1 f32; weights in LDS ----
__global__ void __launch_bounds__(256) k_proj(const float* __restrict__ x,
                                              const float* __restrict__ bproj,
                                              char* __restrict__ ws) {
  extern __shared__ __align__(16) char smem[];
  _Float16* wLds = (_Float16*)smem;                 // [96][96] f16
  if (threadIdx.x < 32) {
    tdm_load_panel_f16(smem, ws + OFF_WPROJT, 96, 96);
    __builtin_amdgcn_s_wait_tensorcnt(0);
  }
  __syncthreads();
  const _Float16* ao = (const _Float16*)(ws + OFF_AO);
  float* x1 = (float*)(ws + OFF_X1);
  int wave = (blockIdx.x * blockDim.x + threadIdx.x) >> 5;
  int lane = threadIdx.x & 31;
  int row0 = wave * 16;
  int hf = lane >> 4, col = lane & 15;
  v16h a0 = load_a(ao + (size_t)row0 * C + 0, C, lane);
  v16h a1 = load_a(ao + (size_t)row0 * C + 32, C, lane);
  v16h a2 = load_a(ao + (size_t)row0 * C + 64, C, lane);
  for (int nt = 0; nt < 6; ++nt) {
    int n0 = nt * 16;
    v16h b0 = load_b(wLds + (size_t)n0 * C + 0, C, lane);
    v16h b1 = load_b(wLds + (size_t)n0 * C + 32, C, lane);
    v16h b2 = load_b(wLds + (size_t)n0 * C + 64, C, lane);
    v8f acc = {};
    acc = wmma_f16(a0, b0, acc);
    acc = wmma_f16(a1, b1, acc);
    acc = wmma_f16(a2, b2, acc);
    int j = n0 + col;
    float bias = bproj[j];
#pragma unroll
    for (int r = 0; r < 8; ++r) {
      int wr = row0 + r + hf * 8;
      int win = wr >> 6, t = wr & 63;
      int b = win >> 6, wl = win & 63;
      int y = ((wl >> 3) << 3) + (t >> 3);
      int xx = ((wl & 7) << 3) + (t & 7);
      size_t tok = (size_t)b * L + (size_t)y * R + xx;
      x1[tok * C + j] = acc[r] + bias + x[tok * C + j];
    }
  }
}

// ---- kernel 5: LN2 (token-major) -> h2 f16 ----
__global__ void k_ln2(const float* __restrict__ g, const float* __restrict__ bt,
                      char* __restrict__ ws) {
  const float* x1 = (const float*)(ws + OFF_X1);
  _Float16* h2 = (_Float16*)(ws + OFF_H2);
  int wave = (blockIdx.x * blockDim.x + threadIdx.x) >> 5;
  int lane = threadIdx.x & 31;
  if (wave >= MROWS) return;
  const float* xr = x1 + (size_t)wave * C;
  float v0 = xr[lane], v1 = xr[lane + 32], v2 = xr[lane + 64];
  float s = v0 + v1 + v2;
#pragma unroll
  for (int m = 16; m >= 1; m >>= 1) s += __shfl_xor(s, m, 32);
  float mean = s * (1.0f / C);
  float d0 = v0 - mean, d1 = v1 - mean, d2 = v2 - mean;
  float q = d0 * d0 + d1 * d1 + d2 * d2;
#pragma unroll
  for (int m = 16; m >= 1; m >>= 1) q += __shfl_xor(q, m, 32);
  float rstd = rsqrtf(q * (1.0f / C) + 1e-5f);
  _Float16* hr = h2 + (size_t)wave * C;
  hr[lane]      = (_Float16)(d0 * rstd * g[lane]      + bt[lane]);
  hr[lane + 32] = (_Float16)(d1 * rstd * g[lane + 32] + bt[lane + 32]);
  hr[lane + 64] = (_Float16)(d2 * rstd * g[lane + 64] + bt[lane + 64]);
}

// ---- kernel 6: fused MLP (fc1 -> GELU -> fc2) + residual; both panels in LDS ----
__global__ void __launch_bounds__(256) k_mlp(const float* __restrict__ bfc1,
                                             const float* __restrict__ bfc2,
                                             float* __restrict__ out,
                                             char* __restrict__ ws) {
  __shared__ __align__(16) _Float16 gbuf[8][16 * 32];
  extern __shared__ __align__(16) char smem[];
  _Float16* w1Lds = (_Float16*)smem;                         // [384][96] f16 (73728 B)
  _Float16* w2Lds = (_Float16*)(smem + 384 * 96 * 2);        // [96][384] f16 (73728 B)
  if (threadIdx.x < 32) {                                    // two TDM DMAs, one wait
    tdm_load_panel_f16(w1Lds, ws + OFF_WFC1T, 96, 384);
    tdm_load_panel_f16(w2Lds, ws + OFF_WFC2T, 384, 96);
    __builtin_amdgcn_s_wait_tensorcnt(0);
  }
  __syncthreads();
  const _Float16* h2 = (const _Float16*)(ws + OFF_H2);
  const float* x1 = (const float*)(ws + OFF_X1);
  int wave = (blockIdx.x * blockDim.x + threadIdx.x) >> 5;
  int w = threadIdx.x >> 5, lane = threadIdx.x & 31;
  int hf = lane >> 4, col = lane & 15;
  int row0 = wave * 16;
  v16h a0 = load_a(h2 + (size_t)row0 * C + 0, C, lane);
  v16h a1 = load_a(h2 + (size_t)row0 * C + 32, C, lane);
  v16h a2 = load_a(h2 + (size_t)row0 * C + 64, C, lane);
  v8f acc2[6];
#pragma unroll
  for (int i = 0; i < 6; ++i) acc2[i] = (v8f){};
  for (int hb = 0; hb < 12; ++hb) {       // 12 hidden chunks of 32
#pragma unroll
    for (int sub = 0; sub < 2; ++sub) {
      int n0 = hb * 32 + sub * 16;
      v16h b0 = load_b(w1Lds + (size_t)n0 * C + 0, C, lane);
      v16h b1 = load_b(w1Lds + (size_t)n0 * C + 32, C, lane);
      v16h b2 = load_b(w1Lds + (size_t)n0 * C + 64, C, lane);
      v8f t = {};
      t = wmma_f16(a0, b0, t);
      t = wmma_f16(a1, b1, t);
      t = wmma_f16(a2, b2, t);
      float bias = bfc1[n0 + col];
#pragma unroll
      for (int r = 0; r < 8; ++r) {
        float xg = t[r] + bias;
        float ge = 0.5f * xg * (1.0f + erff(xg * 0.70710678118654752f));  // exact GELU
        gbuf[w][(r + hf * 8) * 32 + sub * 16 + col] = (_Float16)ge;
      }
    }
    __syncthreads();
    v16h ga = load_a(&gbuf[w][0], 32, lane);
#pragma unroll
    for (int nt = 0; nt < 6; ++nt)
      acc2[nt] = wmma_f16(ga, load_b(w2Lds + (size_t)(nt * 16) * HID + hb * 32, HID, lane), acc2[nt]);
    __syncthreads();
  }
#pragma unroll
  for (int nt = 0; nt < 6; ++nt) {
    int j = nt * 16 + col;
    float bias = bfc2[j];
#pragma unroll
    for (int r = 0; r < 8; ++r) {
      size_t tok = (size_t)row0 + r + hf * 8;
      out[tok * C + j] = acc2[nt][r] + bias + x1[tok * C + j];
    }
  }
}

extern "C" void kernel_launch(void* const* d_in, const int* in_sizes, int n_in,
                              void* d_out, int out_size, void* d_ws, size_t ws_size,
                              hipStream_t stream) {
  const float* x     = (const float*)d_in[0];
  const float* wqkv  = (const float*)d_in[1];
  const float* bqkv  = (const float*)d_in[2];
  const float* wlepe = (const float*)d_in[3];
  const float* blepe = (const float*)d_in[4];
  const float* wproj = (const float*)d_in[5];
  const float* bproj = (const float*)d_in[6];
  const float* g1    = (const float*)d_in[7];
  const float* bt1   = (const float*)d_in[8];
  const float* g2    = (const float*)d_in[9];
  const float* bt2   = (const float*)d_in[10];
  const float* wfc1  = (const float*)d_in[11];
  const float* bfc1  = (const float*)d_in[12];
  const float* wfc2  = (const float*)d_in[13];
  const float* bfc2  = (const float*)d_in[14];
  char* ws = (char*)d_ws;
  float* out = (float*)d_out;
  float* outAttn = out + (size_t)MROWS * C;   // attn follows x_out in d_out

  k_cvtw<<<(288 * 96 + 96 * 96 + 384 * 96 + 96 * 384 + 255) / 256, 256, 0, stream>>>(
      wqkv, wproj, wfc1, wfc2, ws);
  k_ln1<<<MROWS / 8, 256, 0, stream>>>(x, g1, bt1, ws);
  k_qkv<<<(MROWS / 16) / 8, 256, 288 * 96 * 2, stream>>>(bqkv, ws);
  k_attn<<<NWIN * NH, 128, 0, stream>>>(wlepe, blepe, outAttn, ws);
  k_proj<<<(MROWS / 16) / 8, 256, 96 * 96 * 2, stream>>>(x, bproj, ws);
  k_ln2<<<MROWS / 8, 256, 0, stream>>>(g2, bt2, ws);
  k_mlp<<<(MROWS / 16) / 8, 256, (384 * 96 + 96 * 384) * 2, stream>>>(bfc1, bfc2, out, ws);
}
